// MultiHeadAttention_17772574671218
// MI455X (gfx1250) — matrix operand order repaired and checked by reading, not compiled, not run on hardware
//
#include <hip/hip_runtime.h>
#include <hip/hip_bf16.h>

// MI455X / gfx1250 multi-head attention: bf16 WMMA (f32 accum) throughout.
// B=16, N=1024, DIM=768, HEADS=12, HD=64.
// v4: fix async-builtin pointee type (V4i* per clang diagnostic); staging via
//     GLOBAL_LOAD_ASYNC_TO_LDS_B128 (ASYNCcnt), softmax reductions via
//     v_permlane16_b32, double-buffered GEMM mainloop.

typedef __bf16 bf16;
typedef __attribute__((ext_vector_type(16))) __bf16 v16bf;
typedef __attribute__((ext_vector_type(8)))  float  v8f;
typedef __attribute__((ext_vector_type(4)))  int    v4i;

#if __has_builtin(__builtin_amdgcn_global_load_async_to_lds_b128)
#define HAVE_ASYNC_LDS 1
#endif
#if __has_builtin(__builtin_amdgcn_permlane16)
#define HAVE_PERMLANE16 1
#endif

__device__ __forceinline__ v8f v8f_zero() {
  v8f z = {0.f, 0.f, 0.f, 0.f, 0.f, 0.f, 0.f, 0.f};
  return z;
}

__device__ __forceinline__ v8f wmma_bf16(v16bf a, v16bf b, v8f c) {
  // D = A(16x32) * B(32x16) + C, f32 accumulate
  return __builtin_amdgcn_wmma_f32_16x16x32_bf16(
      /*neg_a=*/false, a, /*neg_b=*/false, b,
      /*c_mod=*/(short)0, c, /*reuse_a=*/false, /*reuse_b=*/false);
}

// 16B global->LDS copy: async (no VGPR round-trip, ASYNCcnt) when available.
__device__ __forceinline__ void async_cp16(void* lds, const void* g) {
#if defined(HAVE_ASYNC_LDS)
  __builtin_amdgcn_global_load_async_to_lds_b128(
      (__attribute__((address_space(1))) v4i*)(g),
      (__attribute__((address_space(3))) v4i*)(lds), 0, 0);
#else
  *(uint4*)lds = *(const uint4*)g;
#endif
}

__device__ __forceinline__ void wait_async() {
#if defined(HAVE_ASYNC_LDS)
#if __has_builtin(__builtin_amdgcn_s_wait_asynccnt)
  __builtin_amdgcn_s_wait_asynccnt(0);
#else
  asm volatile("s_wait_asynccnt 0x0" ::: "memory");
#endif
#endif
}

// xor-shuffle within 16-lane halves (matches WMMA C-layout row spread).
#if defined(HAVE_PERMLANE16)
__device__ __forceinline__ float pl16(float v, unsigned s0, unsigned s1) {
  union { float f; unsigned u; } a, r;
  a.f = v;
  r.u = __builtin_amdgcn_permlane16(a.u, a.u, s0, s1, false, false);
  return r.f;
}
__device__ __forceinline__ float red16_max(float v) {
  v = fmaxf(v, pl16(v, 0x67452301u, 0xEFCDAB89u));  // xor 1
  v = fmaxf(v, pl16(v, 0x54761032u, 0xDCFE98BAu));  // xor 2
  v = fmaxf(v, pl16(v, 0x32107654u, 0xBA98FEDCu));  // xor 4
  v = fmaxf(v, pl16(v, 0xFEDCBA98u, 0x76543210u));  // xor 8
  return v;
}
__device__ __forceinline__ float red16_sum(float v) {
  v += pl16(v, 0x67452301u, 0xEFCDAB89u);
  v += pl16(v, 0x54761032u, 0xDCFE98BAu);
  v += pl16(v, 0x32107654u, 0xBA98FEDCu);
  v += pl16(v, 0xFEDCBA98u, 0x76543210u);
  return v;
}
#else
__device__ __forceinline__ float red16_max(float v) {
  for (int m = 1; m < 16; m <<= 1) v = fmaxf(v, __shfl_xor(v, m, 32));
  return v;
}
__device__ __forceinline__ float red16_sum(float v) {
  for (int m = 1; m < 16; m <<= 1) v += __shfl_xor(v, m, 32);
  return v;
}
#endif

// A-fragment (16x32, row-major source, 'stride' bf16 elems per row).
// ISA layout: lane<16 -> row=lane, K={0..7,16..23}; lane>=16 -> row=lane-16, K=+8.
__device__ __forceinline__ v16bf ld_frag_a(const bf16* base, int stride) {
  const int lane = threadIdx.x & 31;
  const bf16* p = base + (lane & 15) * stride + ((lane >> 4) << 3);
  union { v16bf v; uint4 q[2]; } u;
  u.q[0] = *(const uint4*)(p);
  u.q[1] = *(const uint4*)(p + 16);
  return u.v;
}

// B-fragment (32x16). Source stored column-contiguous: base[col*stride + k].
// ISA layout: lane<16 -> col=lane, K={0..7,16..23}; lane>=16 -> col=lane-16, K=+8.
__device__ __forceinline__ v16bf ld_frag_b(const bf16* base, int stride) {
  const int lane = threadIdx.x & 31;
  const bf16* p = base + (lane & 15) * stride + ((lane >> 4) << 3);
  union { v16bf v; uint4 q[2]; } u;
  u.q[0] = *(const uint4*)(p);
  u.q[1] = *(const uint4*)(p + 16);
  return u.v;
}

// ---------------------------------------------------------------- conversion
__global__ void cvt_f32_to_bf16(const float* __restrict__ src,
                                bf16* __restrict__ dst, int n) {
  int i = (blockIdx.x * blockDim.x + threadIdx.x) * 4;
  if (i + 3 < n) {
    const float4 v = *(const float4*)(src + i);
    dst[i + 0] = (bf16)v.x;
    dst[i + 1] = (bf16)v.y;
    dst[i + 2] = (bf16)v.z;
    dst[i + 3] = (bf16)v.w;
  } else {
    for (; i < n; ++i) dst[i] = (bf16)src[i];
  }
}

// ---------------------------------------------------------------- GEMM
// C[M,N] = A[M,K] @ W[N,K]^T + bias  (W stored row-major (N,K) == B columns contiguous)
// MODE 0: scatter bf16 output: Q,K -> [b,h,n,64]; V -> transposed [b,h,64,n]
// MODE 1: plain f32 row-major output (final projection)
template <int MODE>
__global__ __launch_bounds__(256) void gemm_bf16(
    const bf16* __restrict__ A, const bf16* __restrict__ W,
    const float* __restrict__ bias,
    bf16* __restrict__ outQ, bf16* __restrict__ outK, bf16* __restrict__ outV,
    float* __restrict__ outF, int M, int Ncols, int K) {
  __shared__ bf16 As[2][128 * 40];  // double-buffered: 128 rows x 32 k (pad 40)
  __shared__ bf16 Bs[2][128 * 40];

  const int tid  = threadIdx.x;
  const int lane = tid & 31;
  const int wave = tid >> 5;       // 8 waves
  const int wm   = wave & 3;       // rows wm*32
  const int wn   = wave >> 2;      // cols wn*64
  const int m0   = blockIdx.x * 128;
  const int n0   = blockIdx.y * 128;

  v8f acc[2][4];
#pragma unroll
  for (int i = 0; i < 2; ++i)
#pragma unroll
    for (int j = 0; j < 4; ++j) acc[i][j] = v8f_zero();

  const int chunk = tid & 3;   // 8-elem k chunk
  const int lrow  = tid >> 2;  // 0..63

  const bf16* gA = A + (size_t)(m0 + lrow) * K + chunk * 8;
  const bf16* gB = W + (size_t)(n0 + lrow) * K + chunk * 8;
  const int ldso = lrow * 40 + chunk * 8;
  const int nk = K >> 5;  // K/32 steps
  int buf = 0;

#if defined(HAVE_ASYNC_LDS)
  // async pipeline: tile kt+1 streams to LDS while tile kt computes
  {
    async_cp16(&As[0][ldso],           gA);
    async_cp16(&As[0][ldso + 64 * 40], gA + (size_t)64 * K);
    async_cp16(&Bs[0][ldso],           gB);
    async_cp16(&Bs[0][ldso + 64 * 40], gB + (size_t)64 * K);
  }
  wait_async();
  __syncthreads();
  for (int kt = 0; kt < nk; ++kt) {
    const bool more = (kt + 1) < nk;
    if (more) {
      const int k0 = (kt + 1) << 5;
      const int nb = buf ^ 1;  // readers of nb drained at the last barrier
      async_cp16(&As[nb][ldso],           gA + k0);
      async_cp16(&As[nb][ldso + 64 * 40], gA + k0 + (size_t)64 * K);
      async_cp16(&Bs[nb][ldso],           gB + k0);
      async_cp16(&Bs[nb][ldso + 64 * 40], gB + k0 + (size_t)64 * K);
    }
    v16bf af[2], bf_[4];
#pragma unroll
    for (int i = 0; i < 2; ++i)
      af[i] = ld_frag_a(&As[buf][(wm * 32 + i * 16) * 40], 40);
#pragma unroll
    for (int j = 0; j < 4; ++j)
      bf_[j] = ld_frag_b(&Bs[buf][(wn * 64 + j * 16) * 40], 40);
#pragma unroll
    for (int i = 0; i < 2; ++i)
#pragma unroll
      for (int j = 0; j < 4; ++j) acc[i][j] = wmma_bf16(af[i], bf_[j], acc[i][j]);
    if (more) {
      wait_async();
      __syncthreads();
      buf ^= 1;
    }
  }
#else
  // register-staged fallback
  uint4 ra0, ra1, rb0, rb1;
  ra0 = *(const uint4*)(gA);
  ra1 = *(const uint4*)(gA + (size_t)64 * K);
  rb0 = *(const uint4*)(gB);
  rb1 = *(const uint4*)(gB + (size_t)64 * K);
  *(uint4*)&As[0][ldso]           = ra0;
  *(uint4*)&As[0][ldso + 64 * 40] = ra1;
  *(uint4*)&Bs[0][ldso]           = rb0;
  *(uint4*)&Bs[0][ldso + 64 * 40] = rb1;
  __syncthreads();
  for (int kt = 0; kt < nk; ++kt) {
    const bool more = (kt + 1) < nk;
    if (more) {
      const int k0 = (kt + 1) << 5;
      ra0 = *(const uint4*)(gA + k0);
      ra1 = *(const uint4*)(gA + k0 + (size_t)64 * K);
      rb0 = *(const uint4*)(gB + k0);
      rb1 = *(const uint4*)(gB + k0 + (size_t)64 * K);
    }
    v16bf af[2], bf_[4];
#pragma unroll
    for (int i = 0; i < 2; ++i)
      af[i] = ld_frag_a(&As[buf][(wm * 32 + i * 16) * 40], 40);
#pragma unroll
    for (int j = 0; j < 4; ++j)
      bf_[j] = ld_frag_b(&Bs[buf][(wn * 64 + j * 16) * 40], 40);
#pragma unroll
    for (int i = 0; i < 2; ++i)
#pragma unroll
      for (int j = 0; j < 4; ++j) acc[i][j] = wmma_bf16(af[i], bf_[j], acc[i][j]);
    if (more) {
      const int nb = buf ^ 1;
      *(uint4*)&As[nb][ldso]           = ra0;
      *(uint4*)&As[nb][ldso + 64 * 40] = ra1;
      *(uint4*)&Bs[nb][ldso]           = rb0;
      *(uint4*)&Bs[nb][ldso + 64 * 40] = rb1;
      __syncthreads();
      buf = nb;
    }
  }
#endif

  // epilogue: C-matrix layout -> lane holds col = lane&15, rows r + 8*(lane>>4)
  const int cl  = lane & 15;
  const int hse = lane >> 4;
#pragma unroll
  for (int i = 0; i < 2; ++i) {
#pragma unroll
    for (int j = 0; j < 4; ++j) {
      const int c  = n0 + wn * 64 + j * 16 + cl;
      const float bv = bias[c];
#pragma unroll
      for (int r = 0; r < 8; ++r) {
        const int m = m0 + wm * 32 + i * 16 + r + 8 * hse;
        const float val = acc[i][j][r] + bv;
        if constexpr (MODE == 0) {
          // c = which*768 + h*64 + d  (matches reference permute/reshape)
          const int which = c / 768;
          const int rem = c - which * 768;
          const int h = rem >> 6, d = rem & 63;
          const int bb = m >> 10, nn = m & 1023;
          const int bhh = bb * 12 + h;
          if (which == 2) {  // V transposed: [b,h,d,n]
            outV[((size_t)bhh * 64 + d) * 1024 + nn] = (bf16)val;
          } else {
            bf16* dst = (which == 0) ? outQ : outK;
            dst[((size_t)bhh * 1024 + nn) * 64 + d] = (bf16)val;
          }
        } else {
          outF[(size_t)m * Ncols + c] = val;
        }
      }
    }
  }
}

// ---------------------------------------------------------------- attention
// grid.x = b*h (192), grid.y = query tile of 128 (8). 256 threads = 8 waves;
// each wave owns a 16-query strip, flash-style over 128-key LDS tiles.
// K layout [b,h,n,64]; V layout [b,h,64,n] (pre-transposed by qkv GEMM).
__global__ __launch_bounds__(256) void flash_attn(
    const bf16* __restrict__ Q, const bf16* __restrict__ Kb,
    const bf16* __restrict__ Vt, bf16* __restrict__ Ob) {
  __shared__ bf16 Ks[128 * 72];   // [key][d], pad 64->72
  __shared__ bf16 Vs[64 * 136];   // [d][key], pad 128->136
  __shared__ bf16 Ps[8][16 * 40]; // per-wave P bounce buffer (16x32, pad 40)

  const int tid  = threadIdx.x;
  const int lane = tid & 31;
  const int wave = tid >> 5;
  const int bh   = blockIdx.x;           // b*12 + h
  const int q0   = blockIdx.y * 128;
  const int b    = bh / 12, h = bh % 12;
  const float scale = rsqrtf(768.0f);    // reference scales by full dim

  const size_t head_off = (size_t)bh * 1024 * 64;

  // load this wave's Q strip (16x64) as two A-fragments, once
  v16bf qf[2];
  {
    const bf16* qbase = Q + head_off + (size_t)(q0 + wave * 16) * 64;
#pragma unroll
    for (int f = 0; f < 2; ++f) {
      const bf16* p = qbase + (size_t)(lane & 15) * 64 + f * 32 + ((lane >> 4) << 3);
      union { v16bf v; uint4 q[2]; } u;
      u.q[0] = *(const uint4*)p;
      u.q[1] = *(const uint4*)(p + 16);
      qf[f] = u.v;
    }
  }

  float mrow[8], lrow[8];
  v8f oacc[4];
#pragma unroll
  for (int r = 0; r < 8; ++r) { mrow[r] = -1e30f; lrow[r] = 0.f; }
#pragma unroll
  for (int d = 0; d < 4; ++d) oacc[d] = v8f_zero();

  for (int kb = 0; kb < 8; ++kb) {  // 8 key blocks of 128
    __syncthreads();
    {
      // K tile: [128 keys][64 d] row-major -> Ks[key][d]
      const bf16* kg = Kb + head_off + (size_t)kb * 128 * 64;
#pragma unroll
      for (int i = 0; i < 4; ++i) {
        const int idx8 = i * 256 + tid;          // 1024 vec8 chunks
        const int key  = idx8 >> 3;
        const int d0   = (idx8 & 7) * 8;
        async_cp16(&Ks[key * 72 + d0], kg + (size_t)key * 64 + d0);
      }
      // V tile: already transposed in memory [64 d][1024 n] -> Vs[d][key]
      const bf16* vg = Vt + head_off + (size_t)kb * 128;
#pragma unroll
      for (int i = 0; i < 4; ++i) {
        const int idx8 = i * 256 + tid;          // 1024 vec8 chunks over 64x128
        const int d    = idx8 >> 4;              // 16 chunks per d row
        const int kc   = (idx8 & 15) * 8;
        async_cp16(&Vs[d * 136 + kc], vg + (size_t)d * 1024 + kc);
      }
      if (kb + 1 < 8) {  // prefetch next block (global_prefetch_b8)
        __builtin_prefetch(Kb + head_off + (size_t)(kb + 1) * 128 * 64 + tid * 32, 0, 3);
        __builtin_prefetch(Vt + head_off + (size_t)(kb + 1) * 128 + (tid >> 2) * 1024 + (tid & 3) * 32, 0, 3);
      }
    }
    wait_async();
    __syncthreads();

#pragma unroll
    for (int sb = 0; sb < 4; ++sb) {  // 32-key sub-blocks
      const int key0 = sb * 32;

      // S = scale * Q Kt : two 16x16 tiles, each K-dim 64 -> 2 WMMA
      v8f s[2];
#pragma unroll
      for (int t = 0; t < 2; ++t) {
        v8f a = v8f_zero();
        a = wmma_bf16(qf[0], ld_frag_b(&Ks[(key0 + t * 16) * 72 + 0], 72), a);
        a = wmma_bf16(qf[1], ld_frag_b(&Ks[(key0 + t * 16) * 72 + 32], 72), a);
#pragma unroll
        for (int r = 0; r < 8; ++r) a[r] *= scale;
        s[t] = a;
      }

      // online softmax: row reductions via permlane16 within 16-lane halves
      float alpha[8];
#pragma unroll
      for (int r = 0; r < 8; ++r) {
        const float pm = red16_max(fmaxf(s[0][r], s[1][r]));
        const float mnew = fmaxf(mrow[r], pm);
        alpha[r] = __expf(mrow[r] - mnew);
        mrow[r] = mnew;
        const float e0 = __expf(s[0][r] - mnew);
        const float e1 = __expf(s[1][r] - mnew);
        s[0][r] = e0; s[1][r] = e1;
        const float rs = red16_sum(e0 + e1);
        lrow[r] = lrow[r] * alpha[r] + rs;
      }
#pragma unroll
      for (int d = 0; d < 4; ++d)
#pragma unroll
        for (int r = 0; r < 8; ++r) oacc[d][r] *= alpha[r];

      // P (C-layout) -> per-wave LDS -> A-fragment (16x32); DS ops are in-order per wave
      bf16* Pw = Ps[wave];
#pragma unroll
      for (int t = 0; t < 2; ++t)
#pragma unroll
        for (int r = 0; r < 8; ++r)
          Pw[(r + 8 * (lane >> 4)) * 40 + t * 16 + (lane & 15)] = (bf16)s[t][r];
      const v16bf ap = ld_frag_a(Pw, 40);

      // O += P(16x32) @ V(32x64): 4 d-tiles
#pragma unroll
      for (int dt = 0; dt < 4; ++dt) {
        const v16bf bv = ld_frag_b(&Vs[(dt * 16) * 136 + key0], 136);
        oacc[dt] = wmma_bf16(ap, bv, oacc[dt]);
      }
    }
  }

  // normalize and write O in [b, n, h*64+d] bf16 layout for the out-projection
#pragma unroll
  for (int r = 0; r < 8; ++r) {
    const float inv = 1.0f / lrow[r];
    const int row = q0 + wave * 16 + r + 8 * (lane >> 4);
#pragma unroll
    for (int dt = 0; dt < 4; ++dt) {
      const int col = h * 64 + dt * 16 + (lane & 15);
      Ob[((size_t)b * 1024 + row) * 768 + col] = (bf16)(oacc[dt][r] * inv);
    }
  }
}

// ---------------------------------------------------------------- launcher
extern "C" void kernel_launch(void* const* d_in, const int* in_sizes, int n_in,
                              void* d_out, int out_size, void* d_ws, size_t ws_size,
                              hipStream_t stream) {
  const float* x    = (const float*)d_in[0];  // (16,1024,768)
  const float* Wqkv = (const float*)d_in[1];  // (2304,768)
  const float* bqkv = (const float*)d_in[2];  // (2304,)
  const float* Wout = (const float*)d_in[3];  // (768,768)
  const float* bout = (const float*)d_in[4];  // (768,)
  (void)in_sizes; (void)n_in; (void)out_size; (void)ws_size;

  char* ws = (char*)d_ws;
  bf16* xb    = (bf16*)(ws);              // 16384x768       25,165,824 B
  bf16* Wqkvb = (bf16*)(ws + 25165824);   // 2304x768         3,538,944 B
  bf16* Woutb = (bf16*)(ws + 28704768);   // 768x768          1,179,648 B
  bf16* Qb    = (bf16*)(ws + 29884416);   // [b,h,n,64]      25,165,824 B
  bf16* Kbuf  = (bf16*)(ws + 55050240);   // [b,h,n,64]
  bf16* Vtbuf = (bf16*)(ws + 80216064);   // [b,h,64,n] (transposed)
  bf16* Ob    = (bf16*)(ws + 105381888);  // 16384x768 bf16

  cvt_f32_to_bf16<<<12582912 / 1024, 256, 0, stream>>>(x, xb, 12582912);
  cvt_f32_to_bf16<<<1769472 / 1024, 256, 0, stream>>>(Wqkv, Wqkvb, 1769472);
  cvt_f32_to_bf16<<<589824 / 1024, 256, 0, stream>>>(Wout, Woutb, 589824);

  // qkv projection: [16384,768] @ [768,2304] -> scatter Q/K/V (V transposed)
  gemm_bf16<0><<<dim3(128, 18), 256, 0, stream>>>(
      xb, Wqkvb, bqkv, Qb, Kbuf, Vtbuf, nullptr, 16384, 2304, 768);

  // flash attention per (b,h) x query tile
  flash_attn<<<dim3(192, 8), 256, 0, stream>>>(Qb, Kbuf, Vtbuf, Ob);

  // out projection: [16384,768] @ [768,768] + b_out -> f32
  gemm_bf16<1><<<dim3(128, 6), 256, 0, stream>>>(
      Ob, Woutb, bout, nullptr, nullptr, nullptr, (float*)d_out, 16384, 768, 768);
}
